// MoE_40269613367776
// MI455X (gfx1250) — matrix operand order repaired and checked by reading, not compiled, not run on hardware
//
#include <hip/hip_runtime.h>
#include <hip/hip_bf16.h>
#include <stdint.h>
#include <stddef.h>

// ---------------------------------------------------------------------------
// MoE: shared FFN + top-1 routed FFN, bf16 WMMA + TDM async staging (gfx1250)
// Wave tile 32Mx128N (16 WMMA / K-step), block tile 256Mx128N, K double-buffered
// ---------------------------------------------------------------------------

#define N_TOK 16384
#define DDIM  1024
#define HDIM  4096
#define NEXP  8
#define KTILES (DDIM / 32)
#define LDS_BSTRIDE 40   // padded B row stride in bf16 elems (80B, 16B-aligned)

typedef __attribute__((ext_vector_type(16))) __bf16 v16bf;
typedef __attribute__((ext_vector_type(8)))  __bf16 bf16x8;
typedef __attribute__((ext_vector_type(4)))  __bf16 bf16x4;
typedef __attribute__((ext_vector_type(8)))  float  v8f;
typedef __attribute__((ext_vector_type(4)))  float  f32x4;
typedef __attribute__((ext_vector_type(4)))  unsigned int v4u;
typedef __attribute__((ext_vector_type(8)))  int    v8i;
typedef __attribute__((ext_vector_type(4)))  int    v4i;

#if defined(__has_builtin)
#if __has_builtin(__builtin_amdgcn_tensor_load_to_lds) && \
    __has_builtin(__builtin_amdgcn_s_wait_tensorcnt)
#define USE_TDM 1
#endif
#endif

// round-to-nearest-even f32 -> bf16
__device__ __forceinline__ __bf16 f2bf(float f) {
    union { float f; uint32_t u; } a; a.f = f;
    uint32_t r = a.u + 0x7FFFu + ((a.u >> 16) & 1u);
    union { uint16_t u; __bf16 b; } o; o.u = (uint16_t)(r >> 16);
    return o.b;
}

// ---------------------------------------------------------------------------
// 1) f32 -> bf16 conversion
// ---------------------------------------------------------------------------
__global__ void cvt_f32_to_bf16(const float* __restrict__ in,
                                __bf16* __restrict__ out, int n) {
    int i = (blockIdx.x * blockDim.x + threadIdx.x) * 4;
    if (i + 3 < n) {
        f32x4 v = *(const f32x4*)(in + i);
        bf16x4 o;
        o.x = f2bf(v.x); o.y = f2bf(v.y); o.z = f2bf(v.z); o.w = f2bf(v.w);
        *(bf16x4*)(out + i) = o;
    } else {
        for (; i < n; ++i) out[i] = f2bf(in[i]);
    }
}

__global__ void zero_counts(int* __restrict__ counts) {
    if (threadIdx.x < NEXP) counts[threadIdx.x] = 0;
}

// ---------------------------------------------------------------------------
// 2) Router: top-1 argmax of x @ Wr^T + br, bucket tokens per expert.
// ---------------------------------------------------------------------------
__global__ __launch_bounds__(256) void router_kernel(
        const float* __restrict__ x, const float* __restrict__ Wr,
        const float* __restrict__ br, int* __restrict__ counts,
        int* __restrict__ perm) {
    const int tid  = threadIdx.x;
    const int lane = tid & 31;
    const int wv   = tid >> 5;
    const int token = blockIdx.x * 8 + wv;

    float acc[NEXP];
#pragma unroll
    for (int e = 0; e < NEXP; ++e) acc[e] = 0.0f;

    const float* xr = x + (size_t)token * DDIM;
    for (int k = lane; k < DDIM; k += 32) {
        float xv = xr[k];
#pragma unroll
        for (int e = 0; e < NEXP; ++e) acc[e] += xv * Wr[e * DDIM + k];
    }
#pragma unroll
    for (int e = 0; e < NEXP; ++e)
#pragma unroll
        for (int off = 16; off > 0; off >>= 1)
            acc[e] += __shfl_xor(acc[e], off, 32);

    if (lane == 0) {
        int best = 0;
        float bv = acc[0] + br[0];
#pragma unroll
        for (int e = 1; e < NEXP; ++e) {
            float v = acc[e] + br[e];
            if (v > bv) { bv = v; best = e; }   // first-max tiebreak
        }
        int pos = atomicAdd(&counts[best], 1);
        perm[best * N_TOK + pos] = token;
    }
}

// ---------------------------------------------------------------------------
// One K-step of the wave tile: 2 A fragments x 8 B fragments = 16 WMMAs.
//   A (16x32): lanes 0-15 row M=lane, K 0..15; lanes 16-31 K 16..31
//   B (32x16): lanes 0-15 col N=lane, K 0..15; lanes 16-31 K 16..31
//   C (16x16): VGPR r -> (M=r, N=lane) lo-half; (M=r+8, N=lane-16) hi-half
// ---------------------------------------------------------------------------
__device__ __forceinline__ void compute_step(
        const __bf16* __restrict__ cur,
        const __bf16* __restrict__ arow0, const __bf16* __restrict__ arow1,
        int k, int kh, int nl, v8f c0[8], v8f c1[8]) {
    bf16x8 a0lo = *(const bf16x8*)(arow0 + k + kh * 16);
    bf16x8 a0hi = *(const bf16x8*)(arow0 + k + 8 + kh * 16);
    bf16x8 a1lo = *(const bf16x8*)(arow1 + k + kh * 16);
    bf16x8 a1hi = *(const bf16x8*)(arow1 + k + 8 + kh * 16);
    v16bf a0 = __builtin_shufflevector(a0lo, a0hi,
                0,1,2,3,4,5,6,7,8,9,10,11,12,13,14,15);
    v16bf a1 = __builtin_shufflevector(a1lo, a1hi,
                0,1,2,3,4,5,6,7,8,9,10,11,12,13,14,15);
#pragma unroll
    for (int nt = 0; nt < 8; ++nt) {
        v16bf b = *(const v16bf*)(cur + (nt * 16 + nl) * LDS_BSTRIDE + kh * 16);
        c0[nt] = __builtin_amdgcn_wmma_f32_16x16x32_bf16(
                    false, a0, false, b, (short)0, c0[nt], false, false);
        c1[nt] = __builtin_amdgcn_wmma_f32_16x16x32_bf16(
                    false, a1, false, b, (short)0, c1[nt], false, false);
    }
}

// ---------------------------------------------------------------------------
// K loop, double-buffered LDS B tile, manually unrolled x2 so both halves
// use structurally identical accumulator chains (no phi-copy storms).
// ---------------------------------------------------------------------------
__device__ __forceinline__ void gemm_core(
        const __bf16* __restrict__ arow0,  // per-lane A row (M = wv*32 + nl)
        const __bf16* __restrict__ arow1,  // per-lane A row (M = wv*32 + 16 + nl)
        const __bf16* __restrict__ Wtile,  // &W[hBase * DDIM]
        __bf16* lds0, __bf16* lds1,
        int tid, int wv, int kh, int nl,
        v8f c0[8], v8f c1[8]) {
#if defined(USE_TDM)
    const unsigned lds_lo = (unsigned)(unsigned long long)(const void*)lds0;
    const unsigned lds_hi = (unsigned)(unsigned long long)(const void*)lds1;
    const unsigned long long gbase = (unsigned long long)(const void*)Wtile;

    // D# group1: data_size=2B, pad_enable, pad_interval=16 DW (code 3),
    // pad_amount=4 DW (code 3) -> LDS row stride 80B = LDS_BSTRIDE elems.
    v8i g1;
    g1[0] = (1 << 16) | (1 << 20) | (3 << 22) | (3 << 25);
    g1[1] = (int)((unsigned)DDIM << 16);                // tensor_dim0[15:0]
    g1[2] = (int)((unsigned)HDIM << 16) | (DDIM >> 16); // dim0 hi | dim1 lo
    g1[3] = (32 << 16) | (HDIM >> 16);                  // tile_dim0=32 | dim1 hi
    g1[4] = 128;                                        // tile_dim1=128
    g1[5] = DDIM;                                       // tensor_dim0_stride lo
    g1[6] = 0;
    g1[7] = 0;
    const v4i gz4 = {0, 0, 0, 0};
    const v8i gz8 = {0, 0, 0, 0, 0, 0, 0, 0};

    if (wv == 0) {  // prologue: tile 0 -> lds0
        v4u g0;
        g0[0] = 1u;
        g0[1] = lds_lo;
        g0[2] = (unsigned)gbase;
        g0[3] = (unsigned)((gbase >> 32) & 0x1FFFFFFull) | 0x80000000u; // type=2
        __builtin_amdgcn_tensor_load_to_lds(g0, g1, gz4, gz4, gz8, 0);
        __builtin_amdgcn_s_wait_tensorcnt(0);
    }
    __syncthreads();

    for (int i = 0; i < KTILES / 2; ++i) {
        const int k = i * 64;
        // ---- half A: compute from lds0 (tile 2i), DMA tile 2i+1 -> lds1 ----
        if (wv == 0) {
            unsigned long long ga = gbase + (unsigned long long)(2 * i + 1) * 64;
            v4u g0;
            g0[0] = 1u; g0[1] = lds_hi;
            g0[2] = (unsigned)ga;
            g0[3] = (unsigned)((ga >> 32) & 0x1FFFFFFull) | 0x80000000u;
            __builtin_amdgcn_tensor_load_to_lds(g0, g1, gz4, gz4, gz8, 0);
        }
        compute_step(lds0, arow0, arow1, k, kh, nl, c0, c1);
        if (wv == 0) __builtin_amdgcn_s_wait_tensorcnt(0);
        __syncthreads();
        // ---- half B: compute from lds1 (tile 2i+1), DMA tile 2i+2 -> lds0 ----
        if (wv == 0 && 2 * i + 2 < KTILES) {
            unsigned long long ga = gbase + (unsigned long long)(2 * i + 2) * 64;
            v4u g0;
            g0[0] = 1u; g0[1] = lds_lo;
            g0[2] = (unsigned)ga;
            g0[3] = (unsigned)((ga >> 32) & 0x1FFFFFFull) | 0x80000000u;
            __builtin_amdgcn_tensor_load_to_lds(g0, g1, gz4, gz4, gz8, 0);
        }
        compute_step(lds1, arow0, arow1, k + 32, kh, nl, c0, c1);
        if (wv == 0 && 2 * i + 2 < KTILES) __builtin_amdgcn_s_wait_tensorcnt(0);
        __syncthreads();
    }
#else
    // ---- manual double-buffered staging: global -> reg -> LDS ----
    const int row  = tid >> 1;
    const int part = tid & 1;
    const __bf16* grow = Wtile + (size_t)row * DDIM + part * 16;
    const unsigned soff = row * LDS_BSTRIDE + part * 16;
    v16bf stage = *(const v16bf*)grow;              // tile 0

    for (int i = 0; i < KTILES / 2; ++i) {
        const int k = i * 64;
        *(v16bf*)(lds0 + soff) = stage;
        __syncthreads();
        stage = *(const v16bf*)(grow + (2 * i + 1) * 32);
        compute_step(lds0, arow0, arow1, k, kh, nl, c0, c1);

        *(v16bf*)(lds1 + soff) = stage;
        __syncthreads();
        if (2 * i + 2 < KTILES)
            stage = *(const v16bf*)(grow + (2 * i + 2) * 32);
        compute_step(lds1, arow0, arow1, k + 32, kh, nl, c0, c1);
    }
#endif
}

// ---------------------------------------------------------------------------
// 3) Shared GEMM: out[N,H] = x_bf16 @ Ws_bf16^T + bs
// ---------------------------------------------------------------------------
__global__ __launch_bounds__(256) void gemm_shared(
        const __bf16* __restrict__ X, const __bf16* __restrict__ W,
        const float* __restrict__ bias, float* __restrict__ out) {
    __shared__ __bf16 ldsB0[128 * LDS_BSTRIDE];
    __shared__ __bf16 ldsB1[128 * LDS_BSTRIDE];

    const int tid  = threadIdx.x;
    const int lane = tid & 31;
    const int wv   = tid >> 5;
    const int kh   = lane >> 4;
    const int nl   = lane & 15;
    const int mBase = blockIdx.y * 256 + wv * 32;
    const int hBase = blockIdx.x * 128;

    v8f c0[8], c1[8];
#pragma unroll
    for (int i = 0; i < 8; ++i) {
        c0[i] = (v8f){0.f,0.f,0.f,0.f,0.f,0.f,0.f,0.f};
        c1[i] = (v8f){0.f,0.f,0.f,0.f,0.f,0.f,0.f,0.f};
    }

    const __bf16* arow0 = X + (size_t)(mBase + nl) * DDIM;
    const __bf16* arow1 = X + (size_t)(mBase + 16 + nl) * DDIM;
    const __bf16* Wtile = W + (size_t)hBase * DDIM;

    gemm_core(arow0, arow1, Wtile, ldsB0, ldsB1, tid, wv, kh, nl, c0, c1);

#pragma unroll
    for (int nt = 0; nt < 8; ++nt) {
        int col = hBase + nt * 16 + nl;
        float bv = bias[col];
#pragma unroll
        for (int r = 0; r < 8; ++r) {
            int row0 = mBase + r + kh * 8;
            out[(size_t)row0 * HDIM + col] = c0[nt][r] + bv;
            int row1 = mBase + 16 + r + kh * 8;
            out[(size_t)row1 * HDIM + col] = c1[nt][r] + bv;
        }
    }
}

// ---------------------------------------------------------------------------
// 4) Routed GEMM: gathered rows, out[row] += x@We[e]^T + be[e]
// ---------------------------------------------------------------------------
__global__ __launch_bounds__(256) void gemm_routed(
        const __bf16* __restrict__ X, const __bf16* __restrict__ WeAll,
        const float* __restrict__ be, const int* __restrict__ perm,
        const int* __restrict__ counts, float* __restrict__ out) {
    const int e    = blockIdx.z;
    const int cnt  = counts[e];
    const int tile = blockIdx.y;
    if (tile * 256 >= cnt) return;

    __shared__ __bf16 ldsB0[128 * LDS_BSTRIDE];
    __shared__ __bf16 ldsB1[128 * LDS_BSTRIDE];
    __shared__ int rowIdx[256];

    const int tid  = threadIdx.x;
    const int lane = tid & 31;
    const int wv   = tid >> 5;
    const int kh   = lane >> 4;
    const int nl   = lane & 15;
    const int hBase = blockIdx.x * 128;

    {
        int j = tile * 256 + tid;
        rowIdx[tid] = (j < cnt) ? perm[e * N_TOK + j] : -1;
    }
    __syncthreads();

    const int myRow0 = rowIdx[wv * 32 + nl];
    const int myRow1 = rowIdx[wv * 32 + 16 + nl];
    const __bf16* arow0 = X + (size_t)(myRow0 < 0 ? 0 : myRow0) * DDIM;
    const __bf16* arow1 = X + (size_t)(myRow1 < 0 ? 0 : myRow1) * DDIM;
    const __bf16* Wtile = WeAll + (size_t)e * HDIM * DDIM + (size_t)hBase * DDIM;

    v8f c0[8], c1[8];
#pragma unroll
    for (int i = 0; i < 8; ++i) {
        c0[i] = (v8f){0.f,0.f,0.f,0.f,0.f,0.f,0.f,0.f};
        c1[i] = (v8f){0.f,0.f,0.f,0.f,0.f,0.f,0.f,0.f};
    }

    gemm_core(arow0, arow1, Wtile, ldsB0, ldsB1, tid, wv, kh, nl, c0, c1);

#pragma unroll
    for (int nt = 0; nt < 8; ++nt) {
        int col = hBase + nt * 16 + nl;
        float bv = be[e * HDIM + col];
#pragma unroll
        for (int r = 0; r < 8; ++r) {
            int r0 = rowIdx[wv * 32 + r + kh * 8];
            if (r0 >= 0)
                out[(size_t)r0 * HDIM + col] += c0[nt][r] + bv;
            int r1 = rowIdx[wv * 32 + 16 + r + kh * 8];
            if (r1 >= 0)
                out[(size_t)r1 * HDIM + col] += c1[nt][r] + bv;
        }
    }
}

// ---------------------------------------------------------------------------
// launch
// ---------------------------------------------------------------------------
extern "C" void kernel_launch(void* const* d_in, const int* in_sizes, int n_in,
                              void* d_out, int out_size, void* d_ws, size_t ws_size,
                              hipStream_t stream) {
    const float* x  = (const float*)d_in[0];
    const float* Ws = (const float*)d_in[1];
    const float* bs = (const float*)d_in[2];
    const float* We = (const float*)d_in[3];
    const float* be = (const float*)d_in[4];
    const float* Wr = (const float*)d_in[5];
    const float* br = (const float*)d_in[6];
    float* out = (float*)d_out;

    char* ws = (char*)d_ws;
    const size_t xb_bytes  = (size_t)N_TOK * DDIM * 2;          // 32 MB
    const size_t wsb_bytes = (size_t)HDIM * DDIM * 2;           //  8 MB
    const size_t web_bytes = (size_t)NEXP * HDIM * DDIM * 2;    // 64 MB
    __bf16* xb   = (__bf16*)ws;
    __bf16* Wsb  = (__bf16*)(ws + xb_bytes);
    __bf16* Web  = (__bf16*)(ws + xb_bytes + wsb_bytes);
    int*    cnts = (int*)   (ws + xb_bytes + wsb_bytes + web_bytes);
    int*    perm = (int*)   (ws + xb_bytes + wsb_bytes + web_bytes + 256);

    {   int n = N_TOK * DDIM;
        cvt_f32_to_bf16<<<(n / 4 + 255) / 256, 256, 0, stream>>>(x, xb, n); }
    {   int n = HDIM * DDIM;
        cvt_f32_to_bf16<<<(n / 4 + 255) / 256, 256, 0, stream>>>(Ws, Wsb, n); }
    {   int n = NEXP * HDIM * DDIM;
        cvt_f32_to_bf16<<<(n / 4 + 255) / 256, 256, 0, stream>>>(We, Web, n); }

    zero_counts<<<1, 32, 0, stream>>>(cnts);
    router_kernel<<<N_TOK / 8, 256, 0, stream>>>(x, Wr, br, cnts, perm);

    gemm_shared<<<dim3(HDIM / 128, N_TOK / 256), 256, 0, stream>>>(xb, Wsb, bs, out);
    gemm_routed<<<dim3(HDIM / 128, N_TOK / 256, NEXP), 256, 0, stream>>>(
        xb, Web, be, perm, cnts, out);
}
